// NormPreserveDeformConv_56504589746698
// MI455X (gfx1250) — matrix operand order repaired and checked by reading, not compiled, not run on hardware
//
#include <hip/hip_runtime.h>

// ---------------------------------------------------------------------------
// Deformable conv (DCNv1 fwd) as fused bilinear-sample + implicit GEMM
// using CDNA5 v_wmma_f32_16x16x32_bf16 (wave32).
//
// GEMM view: out[oc, n] = sum_gk W[oc, gk] * samp[gk, n]
//   gk = c*9 + tap (matches weight [Oc][C][3][3] flattening)
//   n  = b*4096 + ho*64 + wo
//
// Tile: BM=256 (all of Oc -> zero duplicated sampling), BN=64 (one image row),
// K-chunk 32, double-buffered LDS, 1 barrier per K-step.
// A tiles kept in "fragment-permuted" row order [0..7,16..23,8..15,24..31] so
// every WMMA fragment is a single contiguous 32B LDS read.
// Weights pre-converted once to bf16 tile-major layout in d_ws (if it fits);
// main-loop A staging is then a pure copy, via global_load_async_to_lds_b128
// when the toolchain exposes it.
// ---------------------------------------------------------------------------

typedef __bf16 bf16;
typedef __attribute__((ext_vector_type(16))) __bf16 v16bf;
typedef __attribute__((ext_vector_type(8)))  __bf16 v8bf;
typedef __attribute__((ext_vector_type(2)))  __bf16 v2bf;
typedef __attribute__((ext_vector_type(8)))  float  v8f;
typedef __attribute__((ext_vector_type(4)))  int    v4i;

#define BN     64
#define KC     32
#define CIN    256
#define OC     256
#define KTAPS  9
#define KTOT   (CIN * KTAPS)       // 2304
#define NSTEPS (KTOT / KC)         // 72
#define HH     64
#define WW     64
#define HWSZ   4096                // 64*64

#if defined(__HIP_DEVICE_COMPILE__) && __has_builtin(__builtin_amdgcn_global_load_async_to_lds_b128)
#define HAVE_ASYNC_LDS 1
#else
#define HAVE_ASYNC_LDS 0
#endif

#if defined(__HIP_DEVICE_COMPILE__) && __has_builtin(__builtin_amdgcn_cvt_pk_bf16_f32)
#define HAVE_PK_BF16 1
#else
#define HAVE_PK_BF16 0
#endif

__device__ __forceinline__ bf16 f2bf(float f) {
    union { float f; unsigned int u; } in; in.f = f;
    unsigned int r = in.u + 0x7FFFu + ((in.u >> 16) & 1u);  // round-to-nearest-even
    union { unsigned short s; bf16 b; } out;
    out.s = (unsigned short)(r >> 16);
    return out.b;
}

__device__ __forceinline__ void store_bf_pair(bf16* p, float v0, float v1) {
#if HAVE_PK_BF16
    *(v2bf*)p = __builtin_amdgcn_cvt_pk_bf16_f32(v0, v1);
#else
    p[0] = f2bf(v0); p[1] = f2bf(v1);
#endif
}

// fragment permutation of K index within a 32-wide row: blocks (0,1,2,3)->(0,2,1,3)
__device__ __host__ __forceinline__ int permk(int k) {
    int blk = k >> 3;
    int pb  = ((blk & 1) << 1) | (blk >> 1);
    return (pb << 3) | (k & 7);
}

// ---------------------------------------------------------------------------
// One-time prep: weight fp32 [Oc][2304] -> bf16 tile-major [72][256][32],
// with the fragment permutation applied within each 32-wide row.
// ---------------------------------------------------------------------------
__global__ __launch_bounds__(256)
void wconv_prep_kernel(const float* __restrict__ wgt, bf16* __restrict__ wbf) {
    int d0  = (blockIdx.x * 256 + threadIdx.x) * 8;   // 8 contiguous src cols
    int kt  = d0 >> 13;                                // /(256*32)
    int rem = d0 & 8191;
    int m   = rem >> 5;
    int kk  = rem & 31;                                // 0,8,16,24
    const float* sp = wgt + m * KTOT + kt * KC + kk;
    float4 f0 = *(const float4*)(sp);
    float4 f1 = *(const float4*)(sp + 4);
    bf16 tmp[8];
    tmp[0] = f2bf(f0.x); tmp[1] = f2bf(f0.y); tmp[2] = f2bf(f0.z); tmp[3] = f2bf(f0.w);
    tmp[4] = f2bf(f1.x); tmp[5] = f2bf(f1.y); tmp[6] = f2bf(f1.z); tmp[7] = f2bf(f1.w);
    // 8-blocks stay intact under permk
    *(v8bf*)(wbf + (kt * OC + m) * KC + permk(kk)) = *(v8bf*)tmp;
}

// ---------------------------------------------------------------------------
// Main fused kernel.  USEW: A tiles come pre-converted/permuted from wbf.
// ---------------------------------------------------------------------------
template <bool USEW>
__global__ __launch_bounds__(256)
void deform_conv_wmma_kernel(const float* __restrict__ x,
                             const float* __restrict__ off,
                             const float* __restrict__ wgt,
                             const bf16*  __restrict__ wbf,
                             const float* __restrict__ bias,
                             float* __restrict__ out) {
    __shared__ __attribute__((aligned(32))) bf16   ldsA[2][OC * KC];   // 2 x 16 KB
    __shared__ __attribute__((aligned(32))) bf16   ldsB[2][BN * KC];   // 2 x  4 KB
    __shared__ __attribute__((aligned(16))) float4 pwt[BN * KTAPS];    // 9 KB corner weights
    __shared__ __attribute__((aligned(16))) int4   pofs[BN * KTAPS];   // 9 KB corner offsets

    const int tid  = threadIdx.x;
    const int lane = tid & 31;
    const int wave = tid >> 5;
    const int wm   = wave & 3;          // 4 wave-rows: 64 M each
    const int wn   = wave >> 2;         // 2 wave-cols: 32 N each

    const int nt    = blockIdx.x;       // 512 N-tiles of 64
    const int b     = nt >> 6;          // batch
    const int sbase = (nt & 63) * BN;   // spatial base inside image (one row)

    const float* xb = x + b * CIN * HWSZ;

    // ---------------- per-tile bilinear precompute: 64 positions x 9 taps ---
    for (int idx = tid; idx < BN * KTAPS; idx += 256) {
        int n   = idx / KTAPS;
        int tap = idx - n * KTAPS;
        int s   = sbase + n;
        int ho  = s >> 6, wo = s & 63;
        const float* ob = off + (b * 18 + tap * 2) * HWSZ + s;
        float dy = ob[0];
        float dx = ob[HWSZ];
        float py = dy + (float)(tap / 3 + ho - 1);
        float px = dx + (float)(tap % 3 + wo - 1);
        float fy0 = floorf(py), fx0 = floorf(px);
        float fy = py - fy0,    fx = px - fx0;
        int iy0 = (int)fy0, ix0 = (int)fx0;
        int iy1 = iy0 + 1,  ix1 = ix0 + 1;
        float vy0 = (iy0 >= 0 && iy0 < HH) ? 1.f : 0.f;
        float vy1 = (iy1 >= 0 && iy1 < HH) ? 1.f : 0.f;
        float vx0 = (ix0 >= 0 && ix0 < WW) ? 1.f : 0.f;
        float vx1 = (ix1 >= 0 && ix1 < WW) ? 1.f : 0.f;
        float4 wv;
        wv.x = (1.f - fy) * (1.f - fx) * vy0 * vx0;
        wv.y = (1.f - fy) * fx         * vy0 * vx1;
        wv.z = fy         * (1.f - fx) * vy1 * vx0;
        wv.w = fy         * fx         * vy1 * vx1;
        int cy0 = min(max(iy0, 0), HH - 1), cy1 = min(max(iy1, 0), HH - 1);
        int cx0 = min(max(ix0, 0), WW - 1), cx1 = min(max(ix1, 0), WW - 1);
        pwt[idx] = wv;
        int4 ao;
        ao.x = cy0 * WW + cx0;
        ao.y = cy0 * WW + cx1;
        ao.z = cy1 * WW + cx0;
        ao.w = cy1 * WW + cx1;
        pofs[idx] = ao;
    }

    // stage mappings
    const int  sn  = tid >> 2;                 // sample position 0..63
    const int  sk0 = (tid & 3) * 8;            // 8-wide K segment
    const float* wrow = wgt + tid * KTOT;      // fallback A source (fp32)
    const bf16*  wtil = wbf + tid * KC;        // prepped A source (bf16 tile-major)

    v8f acc[4][2];
    #pragma unroll
    for (int mf = 0; mf < 4; ++mf)
        #pragma unroll
        for (int nf = 0; nf < 2; ++nf)
            acc[mf][nf] = (v8f){0.f, 0.f, 0.f, 0.f, 0.f, 0.f, 0.f, 0.f};

    __syncthreads();   // precompute visible before first staging

    // ---- staging helper ---------------------------------------------------
    auto stage = [&](int kt, int pb) {
        // A tile: row m = tid, 32 K (fragment-permuted order in LDS)
        bf16* la = &ldsA[pb][tid * KC];
        if constexpr (USEW) {
            const bf16* wp = wtil + kt * (OC * KC);   // 64B contiguous, pre-permuted
#if HAVE_ASYNC_LDS
            typedef __attribute__((address_space(1))) v4i gv4i;
            typedef __attribute__((address_space(3))) v4i lv4i;
            #pragma unroll
            for (int j = 0; j < KC; j += 8)
                __builtin_amdgcn_global_load_async_to_lds_b128(
                    (gv4i*)(v4i*)(void*)const_cast<bf16*>(wp + j),
                    (lv4i*)(v4i*)(void*)(la + j), 0, 0);
#else
            #pragma unroll
            for (int j = 0; j < KC; j += 8)
                *(v8bf*)(la + j) = *(const v8bf*)(wp + j);
#endif
        } else {
            const float* wp = wrow + kt * KC;
            #pragma unroll
            for (int j = 0; j < KC; j += 4) {
                float4 f = *(const float4*)(wp + j);
                bf16* d = la + permk(j);              // 4-groups stay inside 8-blocks
                store_bf_pair(d,     f.x, f.y);
                store_bf_pair(d + 2, f.z, f.w);
            }
        }
        // B tile: n = sn, kk = sk0..sk0+7 (fragment-permuted order, matching A)
        {
            int gk  = kt * KC + sk0;
            int c   = gk / 9;
            int tap = gk - c * 9;
            bf16* lb = &ldsB[pb][sn * KC + permk(sk0)];
            #pragma unroll
            for (int i = 0; i < 8; i += 2) {
                float4 wv0 = pwt[sn * KTAPS + tap];
                int4   ao0 = pofs[sn * KTAPS + tap];
                const float* cp0 = xb + (c << 12);
                float v0 = wv0.x * cp0[ao0.x] + wv0.y * cp0[ao0.y]
                         + wv0.z * cp0[ao0.z] + wv0.w * cp0[ao0.w];
                int t2 = tap + 1;
                c   += (t2 == 9) ? 1 : 0;
                tap  = (t2 == 9) ? 0 : t2;

                float4 wv1 = pwt[sn * KTAPS + tap];
                int4   ao1 = pofs[sn * KTAPS + tap];
                const float* cp1 = xb + (c << 12);
                float v1 = wv1.x * cp1[ao1.x] + wv1.y * cp1[ao1.y]
                         + wv1.z * cp1[ao1.z] + wv1.w * cp1[ao1.w];
                t2 = tap + 1;
                c   += (t2 == 9) ? 1 : 0;
                tap  = (t2 == 9) ? 0 : t2;

                store_bf_pair(lb + i, v0, v1);
            }
        }
    };

    auto stage_fence = [&]() {
        if constexpr (USEW) {
#if HAVE_ASYNC_LDS
            asm volatile("s_wait_asynccnt 0" ::: "memory");
#endif
        }
    };

    stage(0, 0);
    stage_fence();
    __syncthreads();

    const int ml  = lane & 15;
    const int kf  = (lane >> 4) * 16;   // fragment K base (A and B both permuted)

    for (int kt = 0; kt < NSTEPS; ++kt) {
        const int cur = kt & 1;

        // ---- fragments: each one contiguous 32B LDS read ------------------
        v16bf afrag[4], bfrag[2];
        #pragma unroll
        for (int mf = 0; mf < 4; ++mf)
            afrag[mf] = *(const v16bf*)(&ldsA[cur][(wm * 64 + mf * 16 + ml) * KC + kf]);
        #pragma unroll
        for (int nf = 0; nf < 2; ++nf)
            bfrag[nf] = *(const v16bf*)(&ldsB[cur][(wn * 32 + nf * 16 + ml) * KC + kf]);

        // ---- stage next tile into other buffer (overlaps with WMMA) ------
        if (kt + 1 < NSTEPS) {
            __builtin_prefetch(USEW ? (const void*)(wtil + (kt + 1) * (OC * KC))
                                    : (const void*)(wrow + (kt + 1) * KC), 0, 3);
            stage(kt + 1, cur ^ 1);
        }

        // ---- 4x2 WMMA accumulate -----------------------------------------
        #pragma unroll
        for (int mf = 0; mf < 4; ++mf)
            #pragma unroll
            for (int nf = 0; nf < 2; ++nf)
                acc[mf][nf] = __builtin_amdgcn_wmma_f32_16x16x32_bf16(
                    false, afrag[mf], false, bfrag[nf],
                    (short)0, acc[mf][nf], false, false);

        stage_fence();
        __syncthreads();   // next buffer visible; current buffer free
    }

    // ---- store: C/D layout lane(l<16): M=row, N=l ; lane(l>=16): M=8+row --
    const int mh = (lane >> 4) * 8;
    #pragma unroll
    for (int mf = 0; mf < 4; ++mf) {
        int mbase = wm * 64 + mf * 16 + mh;
        float bv[8];
        #pragma unroll
        for (int q = 0; q < 8; ++q) bv[q] = bias[mbase + q];
        #pragma unroll
        for (int nf = 0; nf < 2; ++nf) {
            int s = sbase + wn * 32 + nf * 16 + ml;
            float* op = out + ((b * OC + mbase) << 12) + s;
            #pragma unroll
            for (int q = 0; q < 8; ++q)
                op[q << 12] = acc[mf][nf][q] + bv[q];
        }
    }
}

extern "C" void kernel_launch(void* const* d_in, const int* in_sizes, int n_in,
                              void* d_out, int out_size, void* d_ws, size_t ws_size,
                              hipStream_t stream) {
    const float* x    = (const float*)d_in[0];
    const float* off  = (const float*)d_in[1];
    const float* wgt  = (const float*)d_in[2];
    const float* bias = (const float*)d_in[3];
    float* out        = (float*)d_out;

    const size_t need = (size_t)OC * KTOT * sizeof(unsigned short);  // 1.18 MB bf16
    if (ws_size >= need) {
        bf16* wbf = (bf16*)d_ws;
        // 589824 elems / (256 threads * 8) = 288 blocks
        hipLaunchKernelGGL(wconv_prep_kernel, dim3(288), dim3(256), 0, stream, wgt, wbf);
        hipLaunchKernelGGL((deform_conv_wmma_kernel<true>), dim3(512), dim3(256), 0, stream,
                           x, off, wgt, (const bf16*)wbf, bias, out);
    } else {
        hipLaunchKernelGGL((deform_conv_wmma_kernel<false>), dim3(512), dim3(256), 0, stream,
                           x, off, wgt, (const bf16*)nullptr, bias, out);
    }
}
